// CNN_NLP_49624052138378
// MI455X (gfx1250) — matrix operand order repaired and checked by reading, not compiled, not run on hardware
//
#include <hip/hip_runtime.h>
#include <hip/hip_bf16.h>
#include <stdint.h>

// ---------------- problem constants ----------------
#define B_   128
#define S_   128
#define W_   16
#define V_   128
#define E_   32
#define C_   512           // W_*E_ channels after embedding
#define F_   256           // filters per conv
#define SPAD 136           // S_ + 8 padded rows so padded-M GEMM reads stay in-bounds
#define LDSS 88            // LDS row stride in halves: 176B, 16B-aligned, bank-spreading

typedef _Float16 v8h  __attribute__((ext_vector_type(8)));
typedef _Float16 v16h __attribute__((ext_vector_type(16)));
typedef float    v8f  __attribute__((ext_vector_type(8)));

// ---- CDNA5 async global->LDS copy (ASYNCcnt path), 16B per lane -------------
__device__ __forceinline__ void async_ld_b128(const void* gptr, void* lptr) {
  // low 32 bits of a flat LDS pointer are the LDS byte offset (aperture is in
  // the high bits), which is what VDST of global_load_async_to_lds expects.
  uint32_t l = (uint32_t)(uintptr_t)lptr;
  asm volatile("global_load_async_to_lds_b128 %0, %1, off"
               :: "v"(l), "v"(gptr) : "memory");
}
__device__ __forceinline__ void wait_async0() {
  asm volatile("s_wait_asynccnt 0x0" ::: "memory");
}

// ---------------- zero the 8 pad rows per batch of the f16 embedding buffer ---
__global__ __launch_bounds__(256) void k_padzero(_Float16* __restrict__ emb) {
  int u = blockIdx.x * 256 + threadIdx.x;      // 65536 uint4 chunks total
  int rowid = u >> 6;                          // 0..1023  (128 b * 8 pad rows)
  int q = u & 63;                              // 64 uint4 per 512-half row
  int b = rowid >> 3, srel = rowid & 7;
  uint4 z4 = make_uint4(0u, 0u, 0u, 0u);
  *(uint4*)(emb + ((size_t)(b * SPAD + S_ + srel)) * C_ + q * 8) = z4;
}

// ---------------- one wave per (b,s,w): recover one-hot index, gather emb ----
__global__ __launch_bounds__(256) void k_embed(const float* __restrict__ x,
                                               const float* __restrict__ emb_w,
                                               _Float16* __restrict__ emb) {
  int wave = (blockIdx.x * 256 + threadIdx.x) >> 5;   // position id, 262144 total
  int lane = threadIdx.x & 31;
  int w = wave & 15;
  int s = (wave >> 4) & 127;
  int b = wave >> 11;
  // lane reads 4 of the 128 one-hot floats (512B coalesced per wave)
  const float4 xv = *(const float4*)(x + (size_t)wave * V_ + lane * 4);
  float p = xv.x * (float)(lane * 4 + 0) + xv.y * (float)(lane * 4 + 1) +
            xv.z * (float)(lane * 4 + 2) + xv.w * (float)(lane * 4 + 3);
#pragma unroll
  for (int m = 16; m >= 1; m >>= 1) p += __shfl_xor(p, m, 32);
  int idx = (int)(p + 0.5f);
  // E_ == 32 == wave width: lane e writes one f16
  emb[((size_t)(b * SPAD + s)) * C_ + w * E_ + lane] =
      (_Float16)emb_w[idx * E_ + lane];
}

// ---------------- repack conv weights [F][C][k] (f32) -> [F][K=k*512] f16 ----
__global__ __launch_bounds__(256) void k_pack(const float* __restrict__ src,
                                              _Float16* __restrict__ dst,
                                              int k, int Ktot) {
  int d = blockIdx.x * 256 + threadIdx.x;      // 256*Ktot elements
  int f = d / Ktot;
  int kk = d - f * Ktot;
  int dk = kk >> 9;
  int c  = kk & 511;
  dst[d] = (_Float16)src[(f * C_ + c) * k + dk];
}

// ---------------- conv-as-GEMM with WMMA f16 + async double-buffered LDS -----
// A: emb rows (b,t): contiguous K=k*512 window at emb[(b*SPAD+t)*512]
// B: Wt[F][Ktot] (pre-transposed), C: Cout[b][t<128][f] f32
__global__ __launch_bounds__(256) void k_gemm(const _Float16* __restrict__ Aemb,
                                              const _Float16* __restrict__ Wt,
                                              float* __restrict__ Cout,
                                              int Ktot) {
  __shared__ __attribute__((aligned(16))) _Float16 As[2][128 * LDSS];
  __shared__ __attribute__((aligned(16))) _Float16 Bs[2][128 * LDSS];
  const int tid  = threadIdx.x;
  const int lane = tid & 31;
  const int wv   = tid >> 5;          // 8 waves
  const int wmi  = wv >> 1;           // 0..3 : m-tile group
  const int wni  = wv & 1;            // 0..1 : n-tile group
  const int b    = blockIdx.y;        // batch == 128-row M block
  const int bn   = blockIdx.x;        // 0..1 : 128-filter N block
  const int hi   = (lane >= 16) ? 1 : 0;
  const int l15  = lane & 15;

  const _Float16* Abase = Aemb + (size_t)b * (SPAD * C_);
  const _Float16* Bbase = Wt + (size_t)bn * 128 * Ktot;

  // this thread's two 16B staging chunks per matrix per tile
  const int ch0row = tid >> 2,          ch0seg = tid & 3;
  const int ch1row = (tid + 256) >> 2,  ch1seg = (tid + 256) & 3;

  v8f acc[2][4] = {};

  // prologue: async-issue tile 0 into buffer 0
  {
    const int k0 = 0;
    async_ld_b128(Abase + ch0row * C_ + k0 + ch0seg * 8,
                  &As[0][ch0row * LDSS + ch0seg * 8]);
    async_ld_b128(Bbase + (size_t)ch0row * Ktot + k0 + ch0seg * 8,
                  &Bs[0][ch0row * LDSS + ch0seg * 8]);
    async_ld_b128(Abase + ch1row * C_ + k0 + ch1seg * 8,
                  &As[0][ch1row * LDSS + ch1seg * 8]);
    async_ld_b128(Bbase + (size_t)ch1row * Ktot + k0 + ch1seg * 8,
                  &Bs[0][ch1row * LDSS + ch1seg * 8]);
  }

  const int iters = Ktot >> 5;
  for (int kt = 0; kt < iters; ++kt) {
    const int cur = kt & 1;
    const int nxt = cur ^ 1;
    // all of this wave's async loads for tile kt are done (in-order completion)
    wait_async0();
    // now every wave has its share of tile kt in LDS, and everyone is done
    // reading buffer `nxt` from tile kt-1 -> safe to overwrite it
    __syncthreads();

    // async-issue tile kt+1 into the other buffer (branchless; the one-past-end
    // issue reads stay inside the workspace allocation and are never consumed)
    {
      const int k0 = (kt + 1) << 5;
      async_ld_b128(Abase + ch0row * C_ + k0 + ch0seg * 8,
                    &As[nxt][ch0row * LDSS + ch0seg * 8]);
      async_ld_b128(Bbase + (size_t)ch0row * Ktot + k0 + ch0seg * 8,
                    &Bs[nxt][ch0row * LDSS + ch0seg * 8]);
      async_ld_b128(Abase + ch1row * C_ + k0 + ch1seg * 8,
                    &As[nxt][ch1row * LDSS + ch1seg * 8]);
      async_ld_b128(Bbase + (size_t)ch1row * Ktot + k0 + ch1seg * 8,
                    &Bs[nxt][ch1row * LDSS + ch1seg * 8]);
    }

    // A fragments: 16x32 f16 layout — lane<16: K 0..7 & 16..23; lane>=16: +8
    v16h afr[2];
#pragma unroll
    for (int mi = 0; mi < 2; ++mi) {
      int row = (wmi * 2 + mi) * 16 + l15;
      const v8h lo = *(const v8h*)(&As[cur][row * LDSS + hi * 8]);
      const v8h hh = *(const v8h*)(&As[cur][row * LDSS + hi * 8 + 16]);
      v16h a;
#pragma unroll
      for (int i = 0; i < 8; ++i) { a[i] = lo[i]; a[i + 8] = hh[i]; }
      afr[mi] = a;
    }
    // B fragments: 32x16 f16 — lane<16: K 0..7 & 16..23 of col N=lane; lane>=16: +8
    v16h bfr[4];
#pragma unroll
    for (int ni = 0; ni < 4; ++ni) {
      int nrow = (wni * 4 + ni) * 16 + l15;
      const v8h lo = *(const v8h*)(&Bs[cur][nrow * LDSS + hi * 8]);
      const v8h hh = *(const v8h*)(&Bs[cur][nrow * LDSS + hi * 8 + 16]);
      v16h bb;
#pragma unroll
      for (int i = 0; i < 8; ++i) { bb[i] = lo[i]; bb[i + 8] = hh[i]; }
      bfr[ni] = bb;
    }
#pragma unroll
    for (int mi = 0; mi < 2; ++mi)
#pragma unroll
      for (int ni = 0; ni < 4; ++ni)
        acc[mi][ni] = __builtin_amdgcn_wmma_f32_16x16x32_f16(
            false, afr[mi], false, bfr[ni], (short)0, acc[mi][ni], false, false);
  }

  // C 16x16 f32 layout: VGPR r -> M = r + 8*hi, N = l15
#pragma unroll
  for (int mi = 0; mi < 2; ++mi)
#pragma unroll
    for (int ni = 0; ni < 4; ++ni)
#pragma unroll
      for (int r = 0; r < 8; ++r) {
        int t = (wmi * 2 + mi) * 16 + r + hi * 8;
        int f = bn * 128 + (wni * 4 + ni) * 16 + l15;
        Cout[((size_t)b * 128 + t) * F_ + f] = acc[mi][ni][r];
      }
  // trailing speculative async loads are drained by S_ENDPGM's implicit wait
}

// ---------------- deterministic BN stats: per-block partial sums -------------
__global__ __launch_bounds__(256) void k_stats_partial(const float* __restrict__ Cout,
                                                       float* __restrict__ part,
                                                       int Lout) {
  int f = threadIdx.x;
  int blk = blockIdx.x;                    // Lout blocks, 128 valid rows each
  float s = 0.f, ss = 0.f;
  for (int i = 0; i < 128; ++i) {
    int r = blk * 128 + i;
    int b = r / Lout, t = r - b * Lout;
    float v = Cout[((size_t)b * 128 + t) * F_ + f];
    s += v; ss += v * v;
  }
  part[blk * F_ + f] = s;
  part[128 * F_ + blk * F_ + f] = ss;
}

__global__ __launch_bounds__(256) void k_stats_final(const float* __restrict__ part,
                                                     float* __restrict__ musig,
                                                     int Lout) {
  int f = threadIdx.x;
  float s = 0.f, ss = 0.f;
  for (int blk = 0; blk < Lout; ++blk) {
    s  += part[blk * F_ + f];
    ss += part[128 * F_ + blk * F_ + f];
  }
  float inv = 1.0f / (128.0f * (float)Lout);
  float mu  = s * inv;
  float var = ss * inv - mu * mu;          // biased variance (matches reference)
  musig[f]       = mu;
  musig[F_ + f]  = rsqrtf(var + 1e-5f);
}

// ---------------- fused BN * ReLU * global max pool --------------------------
__global__ __launch_bounds__(256) void k_pool(const float* __restrict__ Cout,
                                              const float* __restrict__ musig,
                                              const float* __restrict__ g,
                                              const float* __restrict__ bb,
                                              float* __restrict__ z,
                                              int zoff, int Lout) {
  int f = threadIdx.x;
  int b = blockIdx.x;
  float sc = musig[F_ + f] * g[f];
  float sh = bb[f] - musig[f] * sc;
  float m = -3.4e38f;
  for (int t = 0; t < Lout; ++t) {
    float v = Cout[((size_t)b * 128 + t) * F_ + f];
    m = fmaxf(m, v * sc + sh);
  }
  z[b * 768 + zoff + f] = fmaxf(m, 0.f);   // relu(max) == max(relu)
}

// ---------------- FC1 (bias absorbed by BN1) ---------------------------------
__global__ __launch_bounds__(128) void k_fc1(const float* __restrict__ z,
                                             const float* __restrict__ w,
                                             float* __restrict__ h1) {
  __shared__ float zs[768];
  int b = blockIdx.x, tid = threadIdx.x;
  for (int i = tid; i < 768; i += 128) zs[i] = z[b * 768 + i];
  __syncthreads();
  if (tid < 100) {
    float acc = 0.f;
    for (int kk = 0; kk < 768; ++kk) acc += zs[kk] * w[kk * 100 + tid];
    h1[b * 100 + tid] = acc;
  }
}

// ---------------- BN over batch + ReLU ---------------------------------------
__global__ __launch_bounds__(128) void k_bn1(const float* __restrict__ h1,
                                             const float* __restrict__ g,
                                             const float* __restrict__ bb,
                                             float* __restrict__ h2) {
  int col = blockIdx.x;                    // 100 columns
  int tid = threadIdx.x;                   // batch index
  float v = h1[tid * 100 + col];
  float s = v, ss = v * v;
#pragma unroll
  for (int m = 16; m >= 1; m >>= 1) {
    s  += __shfl_xor(s,  m, 32);
    ss += __shfl_xor(ss, m, 32);
  }
  __shared__ float red[8];
  int wv = tid >> 5;
  if ((tid & 31) == 0) { red[wv] = s; red[4 + wv] = ss; }
  __syncthreads();
  float S  = red[0] + red[1] + red[2] + red[3];
  float SS = red[4] + red[5] + red[6] + red[7];
  float mu  = S / 128.f;
  float var = SS / 128.f - mu * mu;
  float rs  = rsqrtf(var + 1e-5f);
  h2[tid * 100 + col] = fmaxf((v - mu) * rs * g[col] + bb[col], 0.f);
}

// ---------------- FC2 + softmax ---------------------------------------------
__global__ __launch_bounds__(32) void k_fc2(const float* __restrict__ h2,
                                            const float* __restrict__ w,
                                            const float* __restrict__ bias,
                                            float* __restrict__ out) {
  __shared__ float hs[100];
  __shared__ float lg[10];
  int b = blockIdx.x, tid = threadIdx.x;
  for (int i = tid; i < 100; i += 32) hs[i] = h2[b * 100 + i];
  __syncthreads();
  if (tid < 10) {
    float acc = bias[tid];
    for (int kk = 0; kk < 100; ++kk) acc += hs[kk] * w[kk * 10 + tid];
    lg[tid] = acc;
  }
  __syncthreads();
  if (tid < 10) {
    float mx = lg[0];
    for (int i = 1; i < 10; ++i) mx = fmaxf(mx, lg[i]);
    float den = 0.f;
    for (int i = 0; i < 10; ++i) den += __expf(lg[i] - mx);
    out[b * 10 + tid] = __expf(lg[tid] - mx) / den;
  }
}

// ---------------- host-side orchestration ------------------------------------
extern "C" void kernel_launch(void* const* d_in, const int* in_sizes, int n_in,
                              void* d_out, int out_size, void* d_ws, size_t ws_size,
                              hipStream_t stream) {
  (void)in_sizes; (void)n_in; (void)out_size; (void)ws_size;
  const float* x      = (const float*)d_in[0];
  const float* emb_w  = (const float*)d_in[1];
  const float* conv_w[3] = { (const float*)d_in[2], (const float*)d_in[6],
                             (const float*)d_in[10] };
  const float* bn_g[3]   = { (const float*)d_in[4], (const float*)d_in[8],
                             (const float*)d_in[12] };
  const float* bn_b[3]   = { (const float*)d_in[5], (const float*)d_in[9],
                             (const float*)d_in[13] };
  // conv biases d_in[3,7,11] and fc1 bias d_in[15] are absorbed by BN -> unused
  const float* fc1_w = (const float*)d_in[14];
  const float* bn1_g = (const float*)d_in[16];
  const float* bn1_b = (const float*)d_in[17];
  const float* fc2_w = (const float*)d_in[18];
  const float* fc2_b = (const float*)d_in[19];
  float* out = (float*)d_out;

  char* ws = (char*)d_ws;
  _Float16* emb   = (_Float16*)(ws);                       // 17,825,792 B
  _Float16* wt0   = (_Float16*)(ws + 17825792);            //    524,288 B
  _Float16* wt1   = (_Float16*)(ws + 18350080);            //    786,432 B
  _Float16* wt2   = (_Float16*)(ws + 19136512);            //  1,048,576 B
  float*    cvout = (float*)   (ws + 20185088);            // 16,777,216 B
  float*    part  = (float*)   (ws + 36962304);            //    262,144 B
  float*    musig = (float*)   (ws + 37224448);            //      2,048 B
  float*    z     = (float*)   (ws + 37226496);            //    393,216 B
  float*    h1    = (float*)   (ws + 37619712);            //     65,536 B
  float*    h2    = (float*)   (ws + 37685248);            //     65,536 B

  _Float16* wt[3]  = { wt0, wt1, wt2 };
  const int ks[3]   = { 2, 3, 4 };
  const int Ktot[3] = { 1024, 1536, 2048 };
  const int Lout[3] = { 127, 126, 125 };
  const int zoff[3] = { 0, 256, 512 };

  k_padzero<<<256, 256, 0, stream>>>(emb);
  k_embed<<<(B_ * S_ * W_) / 8, 256, 0, stream>>>(x, emb_w, emb);
  for (int i = 0; i < 3; ++i)
    k_pack<<<Ktot[i], 256, 0, stream>>>(conv_w[i], wt[i], ks[i], Ktot[i]);

  for (int i = 0; i < 3; ++i) {
    k_gemm<<<dim3(2, B_), 256, 0, stream>>>(emb, wt[i], cvout, Ktot[i]);
    k_stats_partial<<<Lout[i], 256, 0, stream>>>(cvout, part, Lout[i]);
    k_stats_final<<<1, 256, 0, stream>>>(part, musig, Lout[i]);
    k_pool<<<B_, 256, 0, stream>>>(cvout, musig, bn_g[i], bn_b[i], z, zoff[i],
                                   Lout[i]);
  }

  k_fc1<<<B_, 128, 0, stream>>>(z, fc1_w, h1);
  k_bn1<<<100, 128, 0, stream>>>(h1, bn1_g, bn1_b, h2);
  k_fc2<<<B_, 32, 0, stream>>>(h2, fc2_w, fc2_b, out);
}
